// LangSplatV2Model_85444079386899
// MI455X (gfx1250) — matrix operand order repaired and checked by reading, not compiled, not run on hardware
//
#include <hip/hip_runtime.h>
#include <hip/hip_bf16.h>

// ---------------------------------------------------------------------------
// Problem constants (from the reference)
// ---------------------------------------------------------------------------
#define N_GAUSS   1000000
#define VQ_LAYERS 2
#define CODEBOOK  64
#define KTOT      (VQ_LAYERS * CODEBOOK)   // 128
#define CLIP_D    512
#define M_PIX     (546 * 980)              // 535080
#define C0_CONST  0.28209479177387814f
#define SH0_ELEMS ((size_t)N_GAUSS * KTOT) // 128,000,000 floats

typedef __attribute__((ext_vector_type(16))) __bf16        v16bf;
typedef __attribute__((ext_vector_type(2)))  __bf16        v2bf;
typedef __attribute__((ext_vector_type(8)))  float         v8f;
typedef __attribute__((ext_vector_type(8)))  unsigned int  v8u;

union ABFrag { v8u u; v16bf v; };

// ---------------------------------------------------------------------------
// f32 -> bf16 pair packed into one dword.
// Prefer the native packed convert (v_cvt_pk_bf16_f32) when the toolchain
// exposes it; fall back to native scalar __bf16 casts; last resort: manual RNE.
// ---------------------------------------------------------------------------
#if defined(__has_builtin) && __has_builtin(__builtin_amdgcn_cvt_pk_bf16_f32)
__device__ __forceinline__ unsigned int pk_bf16(float lo, float hi) {
    v2bf r = __builtin_amdgcn_cvt_pk_bf16_f32(lo, hi);
    union { v2bf v; unsigned int u; } cv;
    cv.v = r;
    return cv.u;
}
#elif defined(__gfx1250__)
__device__ __forceinline__ unsigned int pk_bf16(float lo, float hi) {
    union { __bf16 h[2]; unsigned int u; } cv;
    cv.h[0] = (__bf16)lo;
    cv.h[1] = (__bf16)hi;
    return cv.u;
}
#else
__device__ __forceinline__ unsigned int pk_bf16(float lo, float hi) {
    unsigned int a = __float_as_uint(lo);
    unsigned int b = __float_as_uint(hi);
    a = a + 0x7FFFu + ((a >> 16) & 1u);
    b = b + 0x7FFFu + ((b >> 16) & 1u);
    return (a >> 16) | (b & 0xFFFF0000u);
}
#endif

// ---------------------------------------------------------------------------
// Kernel 1: pack codebooks (flat 128 x 512 row-major f32) into per-lane bf16
// WMMA B-fragments in d_ws.
// Fragment index = kc*32 + nt  (kc = K-chunk of 32, nt = N-tile of 16).
// B (32x16, 16-bit) lane layout: lane l holds column n = l&15,
// K = (l>>4)*8 + {0..7,16..23}; dword j packs K = base+kj, base+kj+1 (kj = j<4 ? 2j : 8+2j).
// ---------------------------------------------------------------------------
__global__ void pack_b_kernel(const float* __restrict__ cb,
                              unsigned int* __restrict__ bws) {
    int t    = blockIdx.x * blockDim.x + threadIdx.x;  // 4096 threads total
    int lane = t & 31;
    int frag = t >> 5;                                 // 0..127
    if (frag >= 4 * (CLIP_D / 16)) return;
    int nt    = frag & 31;
    int kc    = frag >> 5;
    int n     = nt * 16 + (lane & 15);
    int kbase = kc * 32 + (lane >> 4) * 8;
    unsigned int* dst = bws + ((size_t)frag * 32 + lane) * 8;
#pragma unroll
    for (int j = 0; j < 8; ++j) {
        int k    = kbase + (j < 4 ? 2 * j : 8 + 2 * j);
        float lo = cb[(size_t)k * CLIP_D + n];
        float hi = cb[(size_t)(k + 1) * CLIP_D + n];
        dst[j]   = pk_bf16(lo, hi);
    }
}

// ---------------------------------------------------------------------------
// Kernel 2: top-4 softmax scatter.  One wave per gaussian row (128 logits).
// Lanes 0-15 handle layer 0 (cols 0..63), lanes 16-31 layer 1 (cols 64..127).
// ---------------------------------------------------------------------------
__device__ __forceinline__ void cswap(float& va, int& ia, float& vb, int& ib) {
    bool keep = (va > vb) || (va == vb && ia < ib);
    if (!keep) {
        float tv = va; va = vb; vb = tv;
        int   ti = ia; ia = ib; ib = ti;
    }
}

__global__ void topk_softmax_kernel(const float* __restrict__ logits,
                                    float* __restrict__ sh0, int nrows) {
    int gw   = (blockIdx.x * blockDim.x + threadIdx.x) >> 5;
    int lane = threadIdx.x & 31;
    if (gw >= nrows) return;

    const float* row = logits + (size_t)gw * KTOT;
    float4 x = *(const float4*)(row + lane * 4);

    int base = (lane & 15) * 4;  // element index within this 64-wide layer
    float v[4] = {x.x, x.y, x.z, x.w};
    int  id[4] = {base, base + 1, base + 2, base + 3};

    // sort 4 descending (index-asc tiebreak) — 5-comparator network
    cswap(v[0], id[0], v[1], id[1]);
    cswap(v[2], id[2], v[3], id[3]);
    cswap(v[0], id[0], v[2], id[2]);
    cswap(v[1], id[1], v[3], id[3]);
    cswap(v[1], id[1], v[2], id[2]);

    // butterfly merge within each 16-lane half: offsets 1,2,4,8
#pragma unroll
    for (int off = 1; off < 16; off <<= 1) {
        float ov[4]; int oid[4];
#pragma unroll
        for (int j = 0; j < 4; ++j) {
            ov[j]  = __shfl_xor(v[j],  off, 32);
            oid[j] = __shfl_xor(id[j], off, 32);
        }
        // merge two sorted-desc 4-lists, keep top 4
        float rv[4]; int rid[4];
        int i = 0, j = 0;
#pragma unroll
        for (int t = 0; t < 4; ++t) {
            bool takeA = (v[i] > ov[j]) || (v[i] == ov[j] && id[i] < oid[j]);
            rv[t]  = takeA ? v[i]  : ov[j];
            rid[t] = takeA ? id[i] : oid[j];
            i += takeA ? 1 : 0;
            j += takeA ? 0 : 1;
        }
#pragma unroll
        for (int t = 0; t < 4; ++t) { v[t] = rv[t]; id[t] = rid[t]; }
    }

    // softmax over the 4 winners (v[0] is the max)
    float e0 = 1.0f;  // exp(v0 - v0)
    float e1 = __expf(v[1] - v[0]);
    float e2 = __expf(v[2] - v[0]);
    float e3 = __expf(v[3] - v[0]);
    float inv = 1.0f / (e0 + e1 + e2 + e3);
    float soft[4] = {e0 * inv, e1 * inv, e2 * inv, e3 * inv};

    const float IC0 = 1.0f / C0_CONST;
    const float NEG = -0.5f * IC0;
    float o[4];
#pragma unroll
    for (int j = 0; j < 4; ++j) {
        float val = NEG;
#pragma unroll
        for (int t = 0; t < 4; ++t)
            if (base + j == id[t]) val = soft[t] * IC0 + NEG;
        o[j] = val;
    }
    *(float4*)(sh0 + (size_t)gw * KTOT + lane * 4) = make_float4(o[0], o[1], o[2], o[3]);
}

// ---------------------------------------------------------------------------
// Kernel 3: WMMA GEMM  (M=535080, K=128, N=512), bf16 inputs, f32 accumulate.
// One block (8 waves) per 16-row M-tile; wave w owns columns [64w, 64w+64).
// A converted f32->bf16 on the fly; B fragments pre-packed in d_ws.
// ---------------------------------------------------------------------------
__global__ void __launch_bounds__(256)
wmma_gemm_kernel(const float* __restrict__ W,
                 const unsigned int* __restrict__ bws,
                 float* __restrict__ out, int M) {
    int wave = threadIdx.x >> 5;       // 0..7
    int lane = threadIdx.x & 31;
    int half = lane >> 4;
    int lm   = lane & 15;
    int mTile = blockIdx.x;

    int m  = mTile * 16 + lm;
    int ml = m < M ? m : (M - 1);      // clamp loads; stores are guarded
    const float* arow = W + (size_t)ml * KTOT;

    v8f acc0 = {}, acc1 = {}, acc2 = {}, acc3 = {};

#pragma unroll
    for (int kc = 0; kc < 4; ++kc) {
        // A fragment (16x32 bf16): half 0 -> K {0..7,16..23}, half 1 -> {8..15,24..31}
        int ks = kc * 32 + half * 8;
        float4 a0 = *(const float4*)(arow + ks);
        float4 a1 = *(const float4*)(arow + ks + 4);
        float4 a2 = *(const float4*)(arow + ks + 16);
        float4 a3 = *(const float4*)(arow + ks + 20);
        ABFrag a;
        a.u[0] = pk_bf16(a0.x, a0.y); a.u[1] = pk_bf16(a0.z, a0.w);
        a.u[2] = pk_bf16(a1.x, a1.y); a.u[3] = pk_bf16(a1.z, a1.w);
        a.u[4] = pk_bf16(a2.x, a2.y); a.u[5] = pk_bf16(a2.z, a2.w);
        a.u[6] = pk_bf16(a3.x, a3.y); a.u[7] = pk_bf16(a3.z, a3.w);

#pragma unroll
        for (int nt = 0; nt < 4; ++nt) {
            int ntG = wave * 4 + nt;
            const unsigned int* bp = bws + (((size_t)kc * 32 + ntG) * 32 + lane) * 8;
            uint4 b0 = *(const uint4*)(bp);
            uint4 b1 = *(const uint4*)(bp + 4);
            ABFrag b;
            b.u[0] = b0.x; b.u[1] = b0.y; b.u[2] = b0.z; b.u[3] = b0.w;
            b.u[4] = b1.x; b.u[5] = b1.y; b.u[6] = b1.z; b.u[7] = b1.w;

            v8f c = (nt == 0) ? acc0 : (nt == 1) ? acc1 : (nt == 2) ? acc2 : acc3;
            c = __builtin_amdgcn_wmma_f32_16x16x32_bf16(
                    /*neg_a=*/false, a.v, /*neg_b=*/false, b.v,
                    /*c_mod=*/(short)0, c, /*reuse_a=*/false, /*reuse_b=*/false);
            if      (nt == 0) acc0 = c;
            else if (nt == 1) acc1 = c;
            else if (nt == 2) acc2 = c;
            else              acc3 = c;
        }
    }

    // C/D layout: VGPR r -> lanes 0-15: (row r, col lane), lanes 16-31: (row r+8, col lane-16)
#pragma unroll
    for (int nt = 0; nt < 4; ++nt) {
        v8f c = (nt == 0) ? acc0 : (nt == 1) ? acc1 : (nt == 2) ? acc2 : acc3;
        int n = wave * 64 + nt * 16 + lm;
#pragma unroll
        for (int r = 0; r < 8; ++r) {
            int row = mTile * 16 + half * 8 + r;
            if (row < M) out[(size_t)row * CLIP_D + n] = c[r];
        }
    }
}

// ---------------------------------------------------------------------------
// Launch
// ---------------------------------------------------------------------------
extern "C" void kernel_launch(void* const* d_in, const int* in_sizes, int n_in,
                              void* d_out, int out_size, void* d_ws, size_t ws_size,
                              hipStream_t stream) {
    const float* logits = (const float*)d_in[0];  // (1e6, 128)
    const float* wmaps  = (const float*)d_in[1];  // (1, 546, 980, 128)
    const float* cbooks = (const float*)d_in[2];  // (2, 64, 512) == flat (128, 512)

    float* sh0  = (float*)d_out;
    float* feat = (float*)d_out + SH0_ELEMS;
    unsigned int* bpack = (unsigned int*)d_ws;    // 128 KB of B fragments

    // 1) pack B fragments: 128 fragments x 32 lanes = 4096 threads
    pack_b_kernel<<<16, 256, 0, stream>>>(cbooks, bpack);

    // 2) top-4 softmax scatter: one wave per row, 8 rows per block
    topk_softmax_kernel<<<(N_GAUSS + 7) / 8, 256, 0, stream>>>(logits, sh0, N_GAUSS);

    // 3) WMMA GEMM: one block per 16-row tile
    wmma_gemm_kernel<<<(M_PIX + 15) / 16, 256, 0, stream>>>(wmaps, bpack, feat, M_PIX);
}
